// SelfAttention_8057358647538
// MI455X (gfx1250) — matrix operand order repaired and checked by reading, not compiled, not run on hardware
//
#include <hip/hip_runtime.h>
#include <hip/hip_bf16.h>

#define BB 4
#define TT 4096
#define DD 1024
#define HH 16
#define DH 64
#define RR (BB * TT)
#define NEGV (-1000000.0f)

typedef __bf16 v16bf __attribute__((ext_vector_type(16)));
typedef float  v8f   __attribute__((ext_vector_type(8)));

union ABu { uint4 u[2]; v16bf v; };

__device__ __forceinline__ unsigned short f2bf(float f) {
  unsigned int u = __float_as_uint(f);
  u += 0x7FFFu + ((u >> 16) & 1u);            // round-to-nearest-even
  return (unsigned short)(u >> 16);
}
__device__ __forceinline__ unsigned int pack2(float a, float b) {
  return (unsigned int)f2bf(a) | ((unsigned int)f2bf(b) << 16);
}
__device__ __forceinline__ v8f wmma_bf16(const ABu& a, const ABu& b, v8f c) {
  // emits v_wmma_f32_16x16x32_bf16
  return __builtin_amdgcn_wmma_f32_16x16x32_bf16(false, a.v, false, b.v, (short)0, c,
                                                 false, false);
}

// ---------------------------------------------------------------------------
// 1) Convert weights to bf16. Wkv = [Wk ; Wv] rows ([N,K]-major), Wp separate.
// ---------------------------------------------------------------------------
__global__ void wconv_kernel(const float* __restrict__ Wk, const float* __restrict__ Wv,
                             const float* __restrict__ Wp,
                             unsigned short* __restrict__ Wkv,
                             unsigned short* __restrict__ Wpb) {
  const size_t DSQ = (size_t)DD * DD;
  size_t i2 = ((size_t)blockIdx.x * blockDim.x + threadIdx.x) * 2;
  if (i2 < 2 * DSQ) {
    const float* src = (i2 < DSQ) ? (Wk + i2) : (Wv + (i2 - DSQ));
    float2 f = *(const float2*)src;
    *(unsigned int*)(Wkv + i2) = pack2(f.x, f.y);
  } else {
    size_t j = i2 - 2 * DSQ;
    float2 f = *(const float2*)(Wp + j);
    *(unsigned int*)(Wpb + j) = pack2(f.x, f.y);
  }
}

// ---------------------------------------------------------------------------
// 2) LayerNorm(x) -> xn bf16 row-major [R, D]. One block (256 thr) per row.
// ---------------------------------------------------------------------------
__global__ void ln_bf16_kernel(const float* __restrict__ x, const float* __restrict__ gamma,
                               const float* __restrict__ beta,
                               unsigned short* __restrict__ xn) {
  __shared__ float s1[256];
  __shared__ float s2[256];
  const int row = blockIdx.x;
  const int tid = threadIdx.x;
  const float* xr = x + (size_t)row * DD;
  float4 v = *(const float4*)(xr + tid * 4);
  s1[tid] = v.x + v.y + v.z + v.w;
  s2[tid] = v.x * v.x + v.y * v.y + v.z * v.z + v.w * v.w;
  __syncthreads();
  for (int st = 128; st > 0; st >>= 1) {
    if (tid < st) { s1[tid] += s1[tid + st]; s2[tid] += s2[tid + st]; }
    __syncthreads();
  }
  const float inv = 1.0f / (float)DD;
  const float mu = s1[0] * inv;
  const float var = s2[0] * inv - mu * mu;
  const float rs = rsqrtf(var + 1e-5f);
  float4 gm = *(const float4*)(gamma + tid * 4);
  float4 bt = *(const float4*)(beta + tid * 4);
  const float y0 = (v.x - mu) * rs * gm.x + bt.x;
  const float y1 = (v.y - mu) * rs * gm.y + bt.y;
  const float y2 = (v.z - mu) * rs * gm.z + bt.z;
  const float y3 = (v.w - mu) * rs * gm.w + bt.w;
  uint2 o;
  o.x = pack2(y0, y1);
  o.y = pack2(y2, y3);
  *(uint2*)(xn + (size_t)row * DD + tid * 4) = o;
}

// ---------------------------------------------------------------------------
// 3) Fused KV GEMM: C[t, n] = xn @ Wkv[n]^T, n in [0,2048).
//    n < 1024 -> keyT f32 [B,H,64,T] (+bk, +mask penalty)
//    n >=1024 -> vT  bf16 [B,H,64,T] (+bv, *mask)
//    Block 128 (4 waves): wave = one M tile, 4 N tiles. grid (256, 32).
// ---------------------------------------------------------------------------
__global__ void kv_gemm_kernel(const unsigned short* __restrict__ A,
                               const unsigned short* __restrict__ Wkv,
                               const float* __restrict__ bk, const float* __restrict__ bv,
                               const float* __restrict__ mask,
                               float* __restrict__ keyT,
                               unsigned short* __restrict__ vT) {
  const int wave = threadIdx.x >> 5;
  const int lane = threadIdx.x & 31;
  const int lm = lane & 15, g = lane >> 4;
  const int mt = blockIdx.x * 4 + wave;   // 0..1023 M tiles over R
  const int ntBase = blockIdx.y * 4;      // N tile group (128 tiles total)
  const int mRow = mt * 16;

  v8f z = {0.f, 0.f, 0.f, 0.f, 0.f, 0.f, 0.f, 0.f};
  v8f acc[4] = {z, z, z, z};
  const unsigned short* aRow = A + (size_t)(mRow + lm) * DD;
  for (int kk = 0; kk < DD; kk += 32) {
    ABu a;
    a.u[0] = *(const uint4*)(aRow + kk + 8 * g);
    a.u[1] = *(const uint4*)(aRow + kk + 16 + 8 * g);
#pragma unroll
    for (int j = 0; j < 4; ++j) {
      const int n = (ntBase + j) * 16 + lm;
      const unsigned short* bRow = Wkv + (size_t)n * DD + kk + 8 * g;
      ABu bb;
      bb.u[0] = *(const uint4*)(bRow);
      bb.u[1] = *(const uint4*)(bRow + 16);
      acc[j] = wmma_bf16(a, bb, acc[j]);
    }
  }
  // Epilogue: transpose-store (C frag rows r are consecutive t -> contiguous).
  const int b = mRow >> 12;                      // T = 4096
  const int tb = (mRow & (TT - 1)) + 8 * g;
  const float* mk = mask + (size_t)b * TT + tb;
  float4 mA = *(const float4*)(mk);
  float4 mB = *(const float4*)(mk + 4);
  float mv[8] = {mA.x, mA.y, mA.z, mA.w, mB.x, mB.y, mB.z, mB.w};
#pragma unroll
  for (int j = 0; j < 4; ++j) {
    const int nt = ntBase + j;
    const int n = nt * 16 + lm;
    if (nt < 64) {  // key path
      const float bias = bk[n];
      const int h = n >> 6, d = n & 63;
      float* o = keyT + (size_t)((b * HH + h) * DH + d) * TT + tb;
      float r0[8];
#pragma unroll
      for (int r = 0; r < 8; ++r) r0[r] = acc[j][r] + bias + (1.0f - mv[r]) * NEGV;
      float4 w0 = {r0[0], r0[1], r0[2], r0[3]};
      float4 w1 = {r0[4], r0[5], r0[6], r0[7]};
      *(float4*)(o) = w0;
      *(float4*)(o + 4) = w1;
    } else {        // v path
      const int n2 = n - DD;
      const float bias = bv[n2];
      const int h = n2 >> 6, l = n2 & 63;
      unsigned short* o = vT + (size_t)((b * HH + h) * DH + l) * TT + tb;
      float r0[8];
#pragma unroll
      for (int r = 0; r < 8; ++r) r0[r] = (acc[j][r] + bias) * mv[r];
      uint4 p;
      p.x = pack2(r0[0], r0[1]);
      p.y = pack2(r0[2], r0[3]);
      p.z = pack2(r0[4], r0[5]);
      p.w = pack2(r0[6], r0[7]);
      *(uint4*)(o) = p;
    }
  }
}

// ---------------------------------------------------------------------------
// 4) k = softmax over T per (b,h,d) row of keyT; emit kT bf16 [B,H,64,T].
//    One block per row; each thread owns 16 contiguous elements.
// ---------------------------------------------------------------------------
__global__ void ksoftmax_kernel(const float* __restrict__ keyT,
                                unsigned short* __restrict__ kT) {
  __shared__ float red[256];
  const int row = blockIdx.x;
  const int tid = threadIdx.x;
  const float* src = keyT + (size_t)row * TT + tid * 16;
  float v[16];
#pragma unroll
  for (int j = 0; j < 4; ++j) {
    float4 t = *(const float4*)(src + j * 4);
    v[4 * j + 0] = t.x; v[4 * j + 1] = t.y; v[4 * j + 2] = t.z; v[4 * j + 3] = t.w;
  }
  float m = v[0];
#pragma unroll
  for (int j = 1; j < 16; ++j) m = fmaxf(m, v[j]);
  red[tid] = m;
  __syncthreads();
  for (int st = 128; st > 0; st >>= 1) {
    if (tid < st) red[tid] = fmaxf(red[tid], red[tid + st]);
    __syncthreads();
  }
  m = red[0];
  __syncthreads();
  float e[16];
  float s = 0.f;
#pragma unroll
  for (int j = 0; j < 16; ++j) { e[j] = __expf(v[j] - m); s += e[j]; }
  red[tid] = s;
  __syncthreads();
  for (int st = 128; st > 0; st >>= 1) {
    if (tid < st) red[tid] += red[tid + st];
    __syncthreads();
  }
  const float invs = 1.0f / red[0];
  uint4 p0, p1;
  p0.x = pack2(e[0] * invs, e[1] * invs);
  p0.y = pack2(e[2] * invs, e[3] * invs);
  p0.z = pack2(e[4] * invs, e[5] * invs);
  p0.w = pack2(e[6] * invs, e[7] * invs);
  p1.x = pack2(e[8] * invs, e[9] * invs);
  p1.y = pack2(e[10] * invs, e[11] * invs);
  p1.z = pack2(e[12] * invs, e[13] * invs);
  p1.w = pack2(e[14] * invs, e[15] * invs);
  unsigned short* dst = kT + (size_t)row * TT + tid * 16;
  *(uint4*)(dst) = p0;
  *(uint4*)(dst + 8) = p1;
}

// ---------------------------------------------------------------------------
// 5) q softmax over DH=64 per (b,t,h); one wave per job; qs bf16 [R, D].
// ---------------------------------------------------------------------------
__global__ void qsoftmax_kernel(const float* __restrict__ query,
                                unsigned short* __restrict__ qs) {
  const int lane = threadIdx.x & 31;
  const int wave = threadIdx.x >> 5;
  const int job = blockIdx.x * 8 + wave;  // R*H jobs
  const int r = job >> 4;
  const int h = job & 15;
  const size_t base = (size_t)r * DD + h * DH + lane * 2;
  float2 qv = *(const float2*)(query + base);
  float m = fmaxf(qv.x, qv.y);
#pragma unroll
  for (int off = 16; off >= 1; off >>= 1) m = fmaxf(m, __shfl_xor(m, off, 32));
  const float e0 = __expf(qv.x - m);
  const float e1 = __expf(qv.y - m);
  float s = e0 + e1;
#pragma unroll
  for (int off = 16; off >= 1; off >>= 1) s += __shfl_xor(s, off, 32);
  const float inv = 1.0f / s;
  *(unsigned int*)(qs + base) = pack2(e0 * inv, e1 * inv);
}

// ---------------------------------------------------------------------------
// 6) attn[b,h] = kT(64xT) @ vT(64xT)^T, K=4096. One block (8 waves) per (b,h);
//    wave owns 2 of 16 tiles sharing the A fragment. Store attnT[l][d] bf16.
// ---------------------------------------------------------------------------
__global__ void attn_kernel(const unsigned short* __restrict__ kT,
                            const unsigned short* __restrict__ vT,
                            unsigned short* __restrict__ attnT) {
  const int bh = blockIdx.x;
  const int wave = threadIdx.x >> 5;
  const int lane = threadIdx.x & 31;
  const int lm = lane & 15, g = lane >> 4;
  const int mt = wave & 3;              // d tile
  const int nt0 = (wave >> 2) * 2;      // l tile pair
  const unsigned short* aRow = kT + ((size_t)bh * DH + mt * 16 + lm) * TT;
  const unsigned short* bRow0 = vT + ((size_t)bh * DH + nt0 * 16 + lm) * TT;
  const unsigned short* bRow1 = bRow0 + (size_t)16 * TT;
  v8f z = {0.f, 0.f, 0.f, 0.f, 0.f, 0.f, 0.f, 0.f};
  v8f acc0 = z, acc1 = z;
  for (int kk = 0; kk < TT; kk += 32) {
    ABu a, b0, b1;
    a.u[0] = *(const uint4*)(aRow + kk + 8 * g);
    a.u[1] = *(const uint4*)(aRow + kk + 16 + 8 * g);
    b0.u[0] = *(const uint4*)(bRow0 + kk + 8 * g);
    b0.u[1] = *(const uint4*)(bRow0 + kk + 16 + 8 * g);
    b1.u[0] = *(const uint4*)(bRow1 + kk + 8 * g);
    b1.u[1] = *(const uint4*)(bRow1 + kk + 16 + 8 * g);
    acc0 = wmma_bf16(a, b0, acc0);
    acc1 = wmma_bf16(a, b1, acc1);
  }
#pragma unroll
  for (int t = 0; t < 2; ++t) {
    const v8f& c = (t == 0) ? acc0 : acc1;
    const int l = (nt0 + t) * 16 + lm;
    unsigned short* o = attnT + ((size_t)bh * DH + l) * DH + mt * 16 + 8 * g;
    uint4 p;
    p.x = pack2(c[0], c[1]);
    p.y = pack2(c[2], c[3]);
    p.z = pack2(c[4], c[5]);
    p.w = pack2(c[6], c[7]);
    *(uint4*)(o) = p;
  }
}

// ---------------------------------------------------------------------------
// 7) y[b,t,h,:] = qs @ attnT^T per (b,h): M=T, N=64, K=64 -> y bf16 [R, D].
//    Block 256 (8 waves), wave = one M tile x 4 N tiles. grid (32, B*H).
// ---------------------------------------------------------------------------
__global__ void y_gemm_kernel(const unsigned short* __restrict__ qs,
                              const unsigned short* __restrict__ attnT,
                              unsigned short* __restrict__ y) {
  const int bh = blockIdx.y;
  const int b = bh >> 4, h = bh & 15;
  const int wave = threadIdx.x >> 5;
  const int lane = threadIdx.x & 31;
  const int lm = lane & 15, g = lane >> 4;
  const int mt = blockIdx.x * 8 + wave;   // 0..255 T tiles
  const int tRow = mt * 16;
  const unsigned short* aRow = qs + (size_t)(b * TT + tRow + lm) * DD + h * DH;
  const unsigned short* Bb = attnT + (size_t)bh * DH * DH;
  v8f z = {0.f, 0.f, 0.f, 0.f, 0.f, 0.f, 0.f, 0.f};
  v8f acc[4] = {z, z, z, z};
#pragma unroll
  for (int kk = 0; kk < DH; kk += 32) {
    ABu a;
    a.u[0] = *(const uint4*)(aRow + kk + 8 * g);
    a.u[1] = *(const uint4*)(aRow + kk + 16 + 8 * g);
#pragma unroll
    for (int j = 0; j < 4; ++j) {
      const int n = j * 16 + lm;
      ABu bb;
      bb.u[0] = *(const uint4*)(Bb + (size_t)n * DH + kk + 8 * g);
      bb.u[1] = *(const uint4*)(Bb + (size_t)n * DH + kk + 16 + 8 * g);
      acc[j] = wmma_bf16(a, bb, acc[j]);
    }
  }
#pragma unroll
  for (int j = 0; j < 4; ++j) {
    const int l = j * 16 + lm;
    unsigned short* o = y + (size_t)(b * TT + tRow + 8 * g) * DD + h * DH + l;
#pragma unroll
    for (int r = 0; r < 8; ++r) o[(size_t)r * DD] = f2bf(acc[j][r]);
  }
}

// ---------------------------------------------------------------------------
// 8) out = y @ Wp^T + bp -> f32 d_out [R, D].
// ---------------------------------------------------------------------------
__global__ void out_gemm_kernel(const unsigned short* __restrict__ y,
                                const unsigned short* __restrict__ Wp,
                                const float* __restrict__ bp,
                                float* __restrict__ out) {
  const int wave = threadIdx.x >> 5;
  const int lane = threadIdx.x & 31;
  const int lm = lane & 15, g = lane >> 4;
  const int mt = blockIdx.x * 4 + wave;   // 1024 M tiles
  const int ntBase = blockIdx.y * 4;      // 64 N tiles / 4
  const int mRow = mt * 16;
  const unsigned short* aRow = y + (size_t)(mRow + lm) * DD;
  v8f z = {0.f, 0.f, 0.f, 0.f, 0.f, 0.f, 0.f, 0.f};
  v8f acc[4] = {z, z, z, z};
  for (int kk = 0; kk < DD; kk += 32) {
    ABu a;
    a.u[0] = *(const uint4*)(aRow + kk + 8 * g);
    a.u[1] = *(const uint4*)(aRow + kk + 16 + 8 * g);
#pragma unroll
    for (int j = 0; j < 4; ++j) {
      const int n = (ntBase + j) * 16 + lm;
      const unsigned short* bRow = Wp + (size_t)n * DD + kk + 8 * g;
      ABu bb;
      bb.u[0] = *(const uint4*)(bRow);
      bb.u[1] = *(const uint4*)(bRow + 16);
      acc[j] = wmma_bf16(a, bb, acc[j]);
    }
  }
#pragma unroll
  for (int j = 0; j < 4; ++j) {
    const int n = (ntBase + j) * 16 + lm;
    const float bias = bp[n];
    float* o = out + (size_t)(mRow + 8 * g) * DD + n;
#pragma unroll
    for (int r = 0; r < 8; ++r) o[(size_t)r * DD] = acc[j][r] + bias;
  }
}

// ---------------------------------------------------------------------------
extern "C" void kernel_launch(void* const* d_in, const int* in_sizes, int n_in,
                              void* d_out, int out_size, void* d_ws, size_t ws_size,
                              hipStream_t stream) {
  (void)in_sizes; (void)n_in; (void)out_size; (void)ws_size;
  const float* query = (const float*)d_in[0];
  const float* x     = (const float*)d_in[1];
  const float* mask  = (const float*)d_in[2];
  const float* gamma = (const float*)d_in[3];
  const float* beta  = (const float*)d_in[4];
  const float* Wk    = (const float*)d_in[5];
  const float* bk    = (const float*)d_in[6];
  const float* Wv    = (const float*)d_in[7];
  const float* bv    = (const float*)d_in[8];
  const float* Wp    = (const float*)d_in[9];
  const float* bp    = (const float*)d_in[10];
  float* out = (float*)d_out;

  char* ws = (char*)d_ws;
  const size_t MB = (size_t)1 << 20;
  unsigned short* xn    = (unsigned short*)(ws + 0);         // 32MB (reused as y)
  float*          keyT  = (float*)(ws + 32 * MB);            // 64MB
  unsigned short* qs    = (unsigned short*)(ws + 32 * MB);   // reuses keyT after use
  unsigned short* vT    = (unsigned short*)(ws + 96 * MB);   // 32MB
  unsigned short* kT    = (unsigned short*)(ws + 128 * MB);  // 32MB
  unsigned short* Wkv   = (unsigned short*)(ws + 160 * MB);  // 4MB
  unsigned short* Wpb   = (unsigned short*)(ws + 164 * MB);  // 2MB
  unsigned short* attnT = (unsigned short*)(ws + 166 * MB);  // 512KB
  unsigned short* yb    = xn;                                // alias (xn dead after KV GEMM)

  wconv_kernel<<<6144, 256, 0, stream>>>(Wk, Wv, Wp, Wkv, Wpb);
  ln_bf16_kernel<<<RR, 256, 0, stream>>>(x, gamma, beta, xn);
  kv_gemm_kernel<<<dim3(256, 32), 128, 0, stream>>>(xn, Wkv, bk, bv, mask, keyT, vT);
  ksoftmax_kernel<<<BB * HH * DH, 256, 0, stream>>>(keyT, kT);
  qsoftmax_kernel<<<RR * HH / 8, 256, 0, stream>>>(query, qs);
  attn_kernel<<<BB * HH, 256, 0, stream>>>(kT, vT, attnT);
  y_gemm_kernel<<<dim3(32, BB * HH), 256, 0, stream>>>(qs, attnT, yb);
  out_gemm_kernel<<<dim3(256, 16), 128, 0, stream>>>(yb, Wpb, bp, out);
}
